// TransformerBlock_12360915878034
// MI455X (gfx1250) — matrix operand order repaired and checked by reading, not compiled, not run on hardware
//
#include <hip/hip_runtime.h>
#include <hip/hip_bf16.h>

// ---------------------------------------------------------------------------
// Transformer block for MI455X (gfx1250, wave32, WMMA).
// Linear attention: ao = q @ (k^T @ v)  (no softmax in reference).
// GEMMs: bf16 A/B in global, staged to LDS (A via GLOBAL_LOAD_ASYNC_TO_LDS_B128,
// B via v_perm transpose), f32 accumulate via v_wmma_f32_16x16x32_bf16.
// ---------------------------------------------------------------------------

#define BQ 4
#define TQ 1024
#define DQ 512
#define HQ 8
#define DFFQ 2048
#define DH 4096      // D*H

#define BM 64
#define BN 128
#define BK 32
#define LDA_S 40     // 80-byte LDS row pitch: 16B aligned, conflict-friendly
#define LDB_S 40

#define FLAG_BIAS   1
#define FLAG_RELU   2
#define FLAG_RES    4
#define FLAG_OUTBF  8

typedef __attribute__((ext_vector_type(16))) __bf16 v16bf;
typedef __attribute__((ext_vector_type(2)))  __bf16 v2bf;
typedef __attribute__((ext_vector_type(8)))  float  v8f;

union Frag16 {
    uint4          q[2];
    unsigned short u[16];
    v16bf          v;
};

// Native hardware f32->bf16 (RNE); pairs fuse to v_cvt_pk_bf16_f32.
__device__ __forceinline__ unsigned short f2bf(float f) {
    __bf16 h = (__bf16)f;
    return __builtin_bit_cast(unsigned short, h);
}
__device__ __forceinline__ unsigned pack2(float a, float b) {
    v2bf p;
    p.x = (__bf16)a;
    p.y = (__bf16)b;
    return __builtin_bit_cast(unsigned, p);
}

// ---------------------------------------------------------------------------
// bf16 WMMA GEMM:  C[z] = op(A[z]) @ B[z] (+bias)(relu)(+res)
//   TA==0 : A is [M x K] row-major bf16          (requires K%32==0)
//   TA==1 : A storage is [K x M] row-major bf16  (computes A^T @ B)
//   B      : [K x N] row-major bf16
// KMASK   : K not a multiple of 32; rows kt..roundup(K,32)-1 must still be
//           readable memory (they are: k/v hold all T rows), values beyond K
//           are zeroed branchlessly.
// Assumes M%64==0, N%128==0 (true for every launch below).
// 256 threads = 8 waves; block tile 64x128; each wave -> 16x64 (4 wmma accs).
// ---------------------------------------------------------------------------
template<int TA, int FLAGS, bool KMASK>
__global__ __launch_bounds__(256) void gemm_bf16_wmma(
    const unsigned short* __restrict__ A, long lda,
    const unsigned short* __restrict__ B, long ldb,
    void* __restrict__ C, long ldc,
    const float* __restrict__ bias,
    const float* __restrict__ res,
    int M, int N, int K,
    long aBO, long aBI, long bBO, long bBI, long cBO, long cBI, int Hdiv)
{
    __shared__ __align__(16) unsigned short As[BM][LDA_S];
    __shared__ __align__(16) unsigned short Bs[BN][LDB_S];

    const int z  = blockIdx.z;
    const int zb = z / Hdiv;
    const int zh = z - zb * Hdiv;
    A += (size_t)zb * aBO + (size_t)zh * aBI;
    B += (size_t)zb * bBO + (size_t)zh * bBI;
    const long coff = (long)zb * cBO + (long)zh * cBI;

    const int m0 = blockIdx.y * BM;
    const int n0 = blockIdx.x * BN;

    const int tid     = threadIdx.x;
    const int lane    = tid & 31;
    const int wv      = tid >> 5;
    const int wm      = (wv >> 1) * 16;   // 0,16,32,48
    const int wn      = (wv & 1) * 64;    // 0,64
    const int halfsel = lane >> 4;        // 0 | 1
    const int lr      = lane & 15;

    v8f acc[4] = {};

    const int Kiter = KMASK ? ((K + BK - 1) & ~(BK - 1)) : K;

    for (int kt = 0; kt < Kiter; kt += BK) {
        // ------------- stage A tile (bf16 copy, K-contiguous) --------------
        if (TA == 0) {
            // Pure byte copy: use the CDNA5 async global->LDS data mover.
            const int row = tid >> 2;            // 0..63
            const int cs  = (tid & 3) * 8;       // 0,8,16,24 (bf16 elems)
            const unsigned short* p = A + (size_t)(m0 + row) * lda + kt + cs;
            const unsigned ldsoff = (unsigned)(uintptr_t)&As[row][cs];
            asm volatile("global_load_async_to_lds_b128 %0, %1, off"
                         :: "v"(ldsoff), "v"(p) : "memory");
            if (kt + BK < Kiter) __builtin_prefetch(p + BK, 0, 1);
        } else {
            // A storage is K x M; transpose while staging: As[m][k] = A[k][m]
            const int ml  = (tid & 15) * 4;      // 0..60
            const int kr  = (tid >> 4) * 2;      // 0..30
            const int k0g = kt + kr, k1g = kt + kr + 1;
            const unsigned short* p0 = A + (size_t)k0g * lda + m0 + ml;
            uint2 q0 = *(const uint2*)p0;        // m: ml..ml+3 @ k0g
            uint2 q1 = *(const uint2*)(p0 + lda);
            if (KMASK) {
                if (!(k0g < K)) { q0.x = 0u; q0.y = 0u; }
                if (!(k1g < K)) { q1.x = 0u; q1.y = 0u; }
            }
            // interleave 16-bit: even/odd m lanes, k pair contiguous
            *(unsigned*)&As[ml + 0][kr] = __builtin_amdgcn_perm(q1.x, q0.x, 0x05040100u);
            *(unsigned*)&As[ml + 1][kr] = __builtin_amdgcn_perm(q1.x, q0.x, 0x07060302u);
            *(unsigned*)&As[ml + 2][kr] = __builtin_amdgcn_perm(q1.y, q0.y, 0x05040100u);
            *(unsigned*)&As[ml + 3][kr] = __builtin_amdgcn_perm(q1.y, q0.y, 0x07060302u);
        }

        // ------------- stage B tile transposed: Bs[n][k] -------------------
        {
            const int nl  = (tid & 15) * 8;      // 0..120
            const int kp  = (tid >> 4) * 2;      // 0..30
            const int k0g = kt + kp, k1g = kt + kp + 1;
            const unsigned short* p0 = B + (size_t)k0g * ldb + n0 + nl;
            uint4 q0 = *(const uint4*)p0;        // n: nl..nl+7 @ k0g
            uint4 q1 = *(const uint4*)(p0 + ldb);
            if (KMASK) {
                if (!(k0g < K)) { q0.x = 0u; q0.y = 0u; q0.z = 0u; q0.w = 0u; }
                if (!(k1g < K)) { q1.x = 0u; q1.y = 0u; q1.z = 0u; q1.w = 0u; }
            }
            if (kt + BK < Kiter) __builtin_prefetch(p0 + (size_t)BK * ldb, 0, 1);
            const unsigned* a0 = (const unsigned*)&q0;
            const unsigned* a1 = (const unsigned*)&q1;
            for (int j = 0; j < 4; ++j) {
                *(unsigned*)&Bs[nl + 2 * j + 0][kp] =
                    __builtin_amdgcn_perm(a1[j], a0[j], 0x05040100u);
                *(unsigned*)&Bs[nl + 2 * j + 1][kp] =
                    __builtin_amdgcn_perm(a1[j], a0[j], 0x07060302u);
            }
        }
        if (TA == 0)
            asm volatile("s_wait_asynccnt 0x0" ::: "memory");
        __syncthreads();

        // ---------------- fragments + WMMA --------------------------------
        // A 16x32 bf16 layout: lanes 0-15 row M=lane, elems 0-7 -> K0..7,
        // elems 8-15 -> K16..23; lanes 16-31 -> K8..15 / K24..31.
        Frag16 afr;
        {
            const unsigned short* arow = &As[wm + lr][0];
            const int ak = halfsel * 8;
            afr.q[0] = *(const uint4*)(arow + ak);
            afr.q[1] = *(const uint4*)(arow + ak + 16);
        }
        // B 32x16 bf16 layout: lanes 0-15 col N=lane, K0..7 / K16..23; lanes 16-31 K8..15 / K24..31.
        for (int jn = 0; jn < 4; ++jn) {
            Frag16 bfr;
            const unsigned short* brow = &Bs[wn + jn * 16 + lr][0];
            const int bk = halfsel * 8;
            bfr.q[0] = *(const uint4*)(brow + bk);
            bfr.q[1] = *(const uint4*)(brow + bk + 16);
            acc[jn] = __builtin_amdgcn_wmma_f32_16x16x32_bf16(
                false, afr.v, false, bfr.v, (short)0, acc[jn], false, false);
        }
        __syncthreads();
    }

    // ---------------- epilogue: bias / relu / residual / store -------------
    for (int jn = 0; jn < 4; ++jn) {
        const int n = n0 + wn + jn * 16 + lr;
        const float bv = (FLAGS & FLAG_BIAS) ? bias[n] : 0.f;
        for (int j = 0; j < 8; ++j) {
            const int m = m0 + wm + j + halfsel * 8;
            float v = acc[jn][j] + bv;
            if (FLAGS & FLAG_RELU) v = fmaxf(v, 0.f);
            if (FLAGS & FLAG_RES)  v += res[(size_t)m * ldc + n];
            if (FLAGS & FLAG_OUTBF)
                ((unsigned short*)C)[coff + (size_t)m * ldc + n] = f2bf(v);
            else
                ((float*)C)[coff + (size_t)m * ldc + n] = v;
        }
    }
}

// ---------------------------------------------------------------------------
// LayerNorm over D=512, one block (128 threads, 4 elems/thread) per row.
// Writes bf16 (packed 8B stores) for the downstream WMMA GEMMs.
// ---------------------------------------------------------------------------
__global__ __launch_bounds__(128) void ln512_kernel(
    const float* __restrict__ x, const float* __restrict__ g,
    const float* __restrict__ be, unsigned short* __restrict__ out)
{
    const int row = blockIdx.x;
    const int t   = threadIdx.x;
    const float* xr = x + (size_t)row * DQ;

    float4 v = *(const float4*)(xr + t * 4);
    __shared__ float red[128];

    red[t] = v.x + v.y + v.z + v.w;
    __syncthreads();
    for (int off = 64; off > 0; off >>= 1) {
        if (t < off) red[t] += red[t + off];
        __syncthreads();
    }
    const float mean = red[0] * (1.f / 512.f);
    __syncthreads();

    float dx = v.x - mean, dy = v.y - mean, dz = v.z - mean, dw = v.w - mean;
    red[t] = dx * dx + dy * dy + dz * dz + dw * dw;
    __syncthreads();
    for (int off = 64; off > 0; off >>= 1) {
        if (t < off) red[t] += red[t + off];
        __syncthreads();
    }
    const float rstd = rsqrtf(red[0] * (1.f / 512.f) + 1e-5f);

    float4 gg = *(const float4*)(g  + t * 4);
    float4 bb = *(const float4*)(be + t * 4);
    float y0 = dx * rstd * gg.x + bb.x;
    float y1 = dy * rstd * gg.y + bb.y;
    float y2 = dz * rstd * gg.z + bb.z;
    float y3 = dw * rstd * gg.w + bb.w;
    *(uint2*)(out + (size_t)row * DQ + t * 4) = make_uint2(pack2(y0, y1), pack2(y2, y3));
}

// ---------------------------------------------------------------------------
// f32 -> bf16 weight conversion (4 elems/thread, packed stores; n % 4 == 0)
// ---------------------------------------------------------------------------
__global__ __launch_bounds__(256) void cvt_f32_bf16(
    const float* __restrict__ s, unsigned short* __restrict__ d, int n)
{
    const int i = (blockIdx.x * blockDim.x + threadIdx.x) * 4;
    if (i < n) {
        float4 v = *(const float4*)(s + i);
        *(uint2*)(d + i) = make_uint2(pack2(v.x, v.y), pack2(v.z, v.w));
    }
}

// ---------------------------------------------------------------------------
extern "C" void kernel_launch(void* const* d_in, const int* in_sizes, int n_in,
                              void* d_out, int out_size, void* d_ws, size_t ws_size,
                              hipStream_t stream)
{
    (void)in_sizes; (void)n_in; (void)out_size; (void)ws_size;

    const float* x     = (const float*)d_in[0];
    const float* Wq    = (const float*)d_in[1];
    const float* bq    = (const float*)d_in[2];
    const float* Wk    = (const float*)d_in[3];
    const float* bk    = (const float*)d_in[4];
    const float* Wv    = (const float*)d_in[5];
    const float* bv    = (const float*)d_in[6];
    const float* Wp    = (const float*)d_in[7];
    const float* bp    = (const float*)d_in[8];
    const float* ln1g  = (const float*)d_in[9];
    const float* ln1b  = (const float*)d_in[10];
    const float* W1    = (const float*)d_in[11];
    const float* b1    = (const float*)d_in[12];
    const float* W2    = (const float*)d_in[13];
    const float* b2    = (const float*)d_in[14];
    const float* ln2g  = (const float*)d_in[15];
    const float* ln2b  = (const float*)d_in[16];
    float* out = (float*)d_out;

    // ---- workspace layout (bytes, 256B aligned) ----
    char* base = (char*)d_ws;
    size_t o = 0;
    auto alloc = [&](size_t bytes) { size_t r = o; o += (bytes + 255) & ~(size_t)255; return r; };
    const size_t W_DxDH = (size_t)DQ * DH;         // 2,097,152
    const size_t W_DxDF = (size_t)DQ * DFFQ;       // 1,048,576
    const size_t ROWS   = (size_t)BQ * TQ;         // 4096
    const size_t oWq  = alloc(W_DxDH * 2);
    const size_t oWk  = alloc(W_DxDH * 2);
    const size_t oWv  = alloc(W_DxDH * 2);
    const size_t oWp  = alloc(W_DxDH * 2);
    const size_t oW1  = alloc(W_DxDF * 2);
    const size_t oW2  = alloc(W_DxDF * 2);
    const size_t oHn  = alloc(ROWS * DQ * 2);      // hn bf16
    const size_t oQ   = alloc(ROWS * DH * 2);      // q  bf16
    const size_t oK   = alloc(ROWS * DH * 2);      // k  bf16
    const size_t oV   = alloc(ROWS * DH * 2);      // v  bf16
    const size_t oM   = alloc((size_t)BQ * HQ * DQ * DQ * 2); // k^T v bf16
    const size_t oAo  = alloc(ROWS * DH * 2);      // ao bf16
    const size_t oX1  = alloc(ROWS * DQ * 4);      // x + proj (f32)
    const size_t oHn2 = alloc(ROWS * DQ * 2);      // hn2 bf16
    const size_t oF1  = alloc(ROWS * DFFQ * 2);    // relu(ffn1) bf16

    unsigned short* Wqb  = (unsigned short*)(base + oWq);
    unsigned short* Wkb  = (unsigned short*)(base + oWk);
    unsigned short* Wvb  = (unsigned short*)(base + oWv);
    unsigned short* Wpb  = (unsigned short*)(base + oWp);
    unsigned short* W1b  = (unsigned short*)(base + oW1);
    unsigned short* W2b  = (unsigned short*)(base + oW2);
    unsigned short* hnb  = (unsigned short*)(base + oHn);
    unsigned short* qb   = (unsigned short*)(base + oQ);
    unsigned short* kb   = (unsigned short*)(base + oK);
    unsigned short* vb   = (unsigned short*)(base + oV);
    unsigned short* Mb   = (unsigned short*)(base + oM);
    unsigned short* aob  = (unsigned short*)(base + oAo);
    float*          x1   = (float*)(base + oX1);
    unsigned short* hn2b = (unsigned short*)(base + oHn2);
    unsigned short* f1b  = (unsigned short*)(base + oF1);

    // ---- weight conversion f32 -> bf16 ----
    auto cvt = [&](const float* s, unsigned short* d, size_t n) {
        cvt_f32_bf16<<<dim3((unsigned)((n / 4 + 255) / 256)), 256, 0, stream>>>(s, d, (int)n);
    };
    cvt(Wq, Wqb, W_DxDH); cvt(Wk, Wkb, W_DxDH); cvt(Wv, Wvb, W_DxDH);
    cvt(Wp, Wpb, W_DxDH); cvt(W1, W1b, W_DxDF); cvt(W2, W2b, W_DxDF);

    // ---- LN1 ----
    ln512_kernel<<<dim3((unsigned)ROWS), 128, 0, stream>>>(x, ln1g, ln1b, hnb);

    // ---- Q, K, V projections: [4096x512] @ [512x4096] + bias -> bf16 ----
    const long ZT = (long)TQ * DH;  // 4,194,304 : per-batch row-block stride
    gemm_bf16_wmma<0, FLAG_BIAS | FLAG_OUTBF, false>
        <<<dim3(DH / BN, (unsigned)(ROWS / BM), 1), 256, 0, stream>>>(
        hnb, DQ, Wqb, DH, qb, DH, bq, nullptr,
        (int)ROWS, DH, DQ, 0, 0, 0, 0, 0, 0, 1);
    gemm_bf16_wmma<0, FLAG_BIAS | FLAG_OUTBF, false>
        <<<dim3(DH / BN, (unsigned)(ROWS / BM), 1), 256, 0, stream>>>(
        hnb, DQ, Wkb, DH, kb, DH, bk, nullptr,
        (int)ROWS, DH, DQ, 0, 0, 0, 0, 0, 0, 1);
    gemm_bf16_wmma<0, FLAG_BIAS | FLAG_OUTBF, false>
        <<<dim3(DH / BN, (unsigned)(ROWS / BM), 1), 256, 0, stream>>>(
        hnb, DQ, Wvb, DH, vb, DH, bv, nullptr,
        (int)ROWS, DH, DQ, 0, 0, 0, 0, 0, 0, 1);

    // ---- M[b,h] = k[b,:T-1,h,:]^T @ v[b,:T-1,h,:]  (512x512, K=1023) ----
    // K=1023: row t=1023 exists in kb/vb (valid memory), masked to zero.
    const long MHD = (long)DQ * DQ;          // 262,144
    gemm_bf16_wmma<1, FLAG_OUTBF, true>
        <<<dim3(DQ / BN, DQ / BM, BQ * HQ), 256, 0, stream>>>(
        kb, DH, vb, DH, Mb, DQ, nullptr, nullptr,
        DQ, DQ, TQ - 1,
        ZT, DQ, ZT, DQ, HQ * MHD, MHD, HQ);

    // ---- ao[b,:,h,:] = q[b,:,h,:] @ M[b,h]  (1024x512, K=512) ----
    gemm_bf16_wmma<0, FLAG_OUTBF, false>
        <<<dim3(DQ / BN, TQ / BM, BQ * HQ), 256, 0, stream>>>(
        qb, DH, Mb, DQ, aob, DH, nullptr, nullptr,
        TQ, DQ, DQ,
        ZT, DQ, HQ * MHD, MHD, ZT, DQ, HQ);

    // ---- x1 = x + ao @ Wp + bp  (f32 out) ----
    gemm_bf16_wmma<0, FLAG_BIAS | FLAG_RES, false>
        <<<dim3(DQ / BN, (unsigned)(ROWS / BM), 1), 256, 0, stream>>>(
        aob, DH, Wpb, DQ, x1, DQ, bp, x,
        (int)ROWS, DQ, DH, 0, 0, 0, 0, 0, 0, 1);

    // ---- LN2 ----
    ln512_kernel<<<dim3((unsigned)ROWS), 128, 0, stream>>>(x1, ln2g, ln2b, hn2b);

    // ---- ffn1 = relu(hn2 @ W1 + b1) -> bf16 ----
    gemm_bf16_wmma<0, FLAG_BIAS | FLAG_RELU | FLAG_OUTBF, false>
        <<<dim3(DFFQ / BN, (unsigned)(ROWS / BM), 1), 256, 0, stream>>>(
        hn2b, DQ, W1b, DFFQ, f1b, DFFQ, b1, nullptr,
        (int)ROWS, DFFQ, DQ, 0, 0, 0, 0, 0, 0, 1);

    // ---- out = x1 + ffn1 @ W2 + b2  (f32 out) ----
    gemm_bf16_wmma<0, FLAG_BIAS | FLAG_RES, false>
        <<<dim3(DQ / BN, (unsigned)(ROWS / BM), 1), 256, 0, stream>>>(
        f1b, DFFQ, W2b, DQ, out, DQ, b2, x1,
        (int)ROWS, DQ, DFFQ, 0, 0, 0, 0, 0, 0, 1);
}
